// HunYuanMoEDecoderLayer_56650618635039
// MI455X (gfx1250) — matrix operand order repaired and checked by reading, not compiled, not run on hardware
//
#include <hip/hip_runtime.h>
#include <cstdint>
#include <cmath>

typedef __bf16 bf16_t;
typedef __attribute__((ext_vector_type(16))) __bf16 v16bf;
typedef __attribute__((ext_vector_type(8)))  __bf16 v8bf;
typedef __attribute__((ext_vector_type(8)))  float  v8f;

#define T_TOK   2048
#define DMODEL  2048
#define NH      16
#define NKV     4
#define HD      128
#define NEXP    16
#define I_RT    1024
#define I_SH    4096

// ---- WMMA GEMM tiling ----
#define BM 128
#define BN 128
#define BK 32
#define LS 40    // padded LDS row stride (elems): 80B -> 16B aligned, spreads banks

// CDNA5 async global->LDS copy (tracked by ASYNCcnt). 16 bytes per lane.
__device__ __forceinline__ void async_ld16(bf16_t* lds, const bf16_t* g) {
  unsigned off = (unsigned)(uintptr_t)lds;   // low 32 bits of generic ptr == LDS addr
  asm volatile("global_load_async_to_lds_b128 %0, %1, off"
               :: "v"(off), "v"(g) : "memory");
}
__device__ __forceinline__ void stage8(bf16_t* lds, const bf16_t* g) {
  async_ld16(lds, g);
}
__device__ __forceinline__ void stage8(bf16_t* lds, const float* g) {
  v8bf o;
#pragma unroll
  for (int i = 0; i < 8; ++i) o[i] = (bf16_t)g[i];
  *(v8bf*)lds = o;
}

// C[m,n] = (acc? C : 0) + alpha * rowscale[m] * sum_k A[m,k] * B[n,k]
// A: bf16 [M,K] row-major (lda). B: TB [N,K] row-major (ldb). C: f32 (ldc).
// 128x128 block tile, 8 wave32s each owning 32x64; double-buffered LDS with
// async-DMA prefetch of tile k+1 overlapped with WMMA on tile k.
template <typename TB>
__global__ __launch_bounds__(256) void gemm_bf16_kernel(
    const bf16_t* __restrict__ A, const TB* __restrict__ B, float* __restrict__ C,
    int M, int N, int K, int lda, int ldb, int ldc,
    float alpha, const float* __restrict__ rowscale, int rs_stride, int accumulate) {
  (void)M; (void)N;
  __shared__ bf16_t As[2][BM * LS];
  __shared__ bf16_t Bs[2][BN * LS];

  const int tid  = threadIdx.x;
  const int lane = tid & 31;
  const int l16  = lane & 15;
  const int hsel = lane >> 4;          // lane half selects K sub-range per ISA layout
  const int wave = tid >> 5;           // 8 wave32s
  const int wm   = (wave & 3) * 32;    // 4x2 wave grid -> 128x128 block tile
  const int wn   = (wave >> 2) * 64;
  const int m0   = blockIdx.y * BM;
  const int n0   = blockIdx.x * BN;

  v8f acc[2][4] = {};

  auto stageA = [&](int buf, int k0) {
#pragma unroll
    for (int j = 0; j < 2; ++j) {
      int e = (tid + j * 256) * 8;
      int r = e >> 5, c = e & 31;
      async_ld16(&As[buf][r * LS + c], A + (size_t)(m0 + r) * lda + k0 + c);
    }
  };
  auto stageB = [&](int buf, int k0) {
#pragma unroll
    for (int j = 0; j < 2; ++j) {
      int e = (tid + j * 256) * 8;
      int r = e >> 5, c = e & 31;
      stage8(&Bs[buf][r * LS + c], B + (size_t)(n0 + r) * ldb + k0 + c);
    }
  };

  stageA(0, 0);
  stageB(0, 0);
  asm volatile("s_wait_asynccnt 0" ::: "memory");
  __syncthreads();

  int buf = 0;
  for (int k0 = 0; k0 < K; k0 += BK) {
    // Issue DMA for next tile into the other buffer; overlaps with WMMA below.
    if (k0 + BK < K) {
      stageA(buf ^ 1, k0 + BK);
      stageB(buf ^ 1, k0 + BK);
      if (k0 + 2 * BK < K) {  // pull k+2 weight slice toward L2
        int e = tid * 8;
        int r = e >> 5, c = e & 31;
        __builtin_prefetch(B + (size_t)(n0 + r) * ldb + k0 + 2 * BK + c, 0, 1);
      }
    }

    // Fragments per ISA 7.12.2 (wave32, 16x16x32 bf16)
    v16bf afrag[2], bfrag[4];
#pragma unroll
    for (int tm = 0; tm < 2; ++tm) {
      const bf16_t* ap = &As[buf][(wm + tm * 16 + l16) * LS + hsel * 8];
      v8bf lo = *(const v8bf*)ap;          // K = kb .. kb+7
      v8bf hi = *(const v8bf*)(ap + 16);   // K = kb+16 .. kb+23
#pragma unroll
      for (int i = 0; i < 8; ++i) { afrag[tm][i] = lo[i]; afrag[tm][i + 8] = hi[i]; }
    }
#pragma unroll
    for (int tn = 0; tn < 4; ++tn) {
      const bf16_t* bp = &Bs[buf][(wn + tn * 16 + l16) * LS + hsel * 8];
      v8bf lo = *(const v8bf*)bp;
      v8bf hi = *(const v8bf*)(bp + 16);
#pragma unroll
      for (int i = 0; i < 8; ++i) { bfrag[tn][i] = lo[i]; bfrag[tn][i + 8] = hi[i]; }
    }
#pragma unroll
    for (int tm = 0; tm < 2; ++tm)
#pragma unroll
      for (int tn = 0; tn < 4; ++tn)
        acc[tm][tn] = __builtin_amdgcn_wmma_f32_16x16x32_bf16(
            false, afrag[tm], false, bfrag[tn], (short)0, acc[tm][tn], false, false);

    // Own async copies into buf^1 must land before the barrier publishes them.
    asm volatile("s_wait_asynccnt 0" ::: "memory");
    __syncthreads();
    buf ^= 1;
  }

  // Epilogue: C layout = lane {0-15:M=v, 16-31:M=8+v}, N = lane&15
#pragma unroll
  for (int tm = 0; tm < 2; ++tm)
#pragma unroll
    for (int tn = 0; tn < 4; ++tn)
#pragma unroll
      for (int v = 0; v < 8; ++v) {
        int m = m0 + wm + tm * 16 + hsel * 8 + v;
        int n = n0 + wn + tn * 16 + l16;
        float s = alpha;
        if (rowscale) s *= rowscale[(size_t)m * rs_stride];
        float val = s * acc[tm][tn][v];
        float* cp = C + (size_t)m * ldc + n;
        if (accumulate) *cp = *cp + val; else *cp = val;
      }
}

// ---- small kernels ----
__global__ __launch_bounds__(256) void copyf_kernel(const float* __restrict__ x,
                                                    float* __restrict__ y, size_t n) {
  size_t i = (size_t)blockIdx.x * 256 + threadIdx.x;
  if (i < n) y[i] = x[i];
}
__global__ __launch_bounds__(256) void cast_bf16_kernel(const float* __restrict__ x,
                                                        bf16_t* __restrict__ y, size_t n) {
  size_t i = (size_t)blockIdx.x * 256 + threadIdx.x;
  if (i < n) y[i] = (bf16_t)x[i];
}
__global__ __launch_bounds__(256) void silu_mul_kernel(const float* __restrict__ g,
                                                       const float* __restrict__ u,
                                                       bf16_t* __restrict__ h, size_t n) {
  size_t i = (size_t)blockIdx.x * 256 + threadIdx.x;
  if (i < n) { float gv = g[i]; h[i] = (bf16_t)(gv / (1.f + __expf(-gv)) * u[i]); }
}
__global__ __launch_bounds__(256) void rmsnorm_bf16_kernel(const float* __restrict__ x,
                                                           const float* __restrict__ w,
                                                           bf16_t* __restrict__ out, int D) {
  __shared__ float red[256];
  int t = blockIdx.x;
  const float* xr = x + (size_t)t * D;
  float ss = 0.f;
  for (int d = threadIdx.x; d < D; d += 256) { float v = xr[d]; ss += v * v; }
  red[threadIdx.x] = ss; __syncthreads();
  for (int s = 128; s > 0; s >>= 1) {
    if (threadIdx.x < s) red[threadIdx.x] += red[threadIdx.x + s];
    __syncthreads();
  }
  float r = rsqrtf(red[0] / (float)D + 1e-6f);
  for (int d = threadIdx.x; d < D; d += 256)
    out[(size_t)t * D + d] = (bf16_t)(w[d] * xr[d] * r);
}
// RoPE (rotate-half) + per-head RMSNorm, fp32 in -> bf16 out. One block per (token, head).
__global__ __launch_bounds__(128) void rope_qknorm_kernel(
    const float* __restrict__ q, const float* __restrict__ k,
    const float* __restrict__ cost, const float* __restrict__ sint,
    const float* __restrict__ qn_w, const float* __restrict__ kn_w,
    bf16_t* __restrict__ qb, bf16_t* __restrict__ kb) {
  __shared__ float buf[HD];
  __shared__ float red[HD];
  int b = blockIdx.x;
  int t = b / (NH + NKV);
  int h = b % (NH + NKV);
  int d = threadIdx.x;
  const float* src; const float* gw; bf16_t* dst;
  if (h < NH) { src = q + ((size_t)t * NH + h) * HD;         gw = qn_w; dst = qb + ((size_t)t * NH + h) * HD; }
  else        { src = k + ((size_t)t * NKV + (h - NH)) * HD; gw = kn_w; dst = kb + ((size_t)t * NKV + (h - NH)) * HD; }
  float x = src[d];
  buf[d] = x; __syncthreads();
  float rh = (d < HD / 2) ? -buf[d + HD / 2] : buf[d - HD / 2];
  float c = cost[(size_t)t * HD + d], sn = sint[(size_t)t * HD + d];  // position == t (arange)
  float v = x * c + rh * sn;
  red[d] = v * v; __syncthreads();
  for (int s = 64; s > 0; s >>= 1) {
    if (d < s) red[d] += red[d + s];
    __syncthreads();
  }
  float r = rsqrtf(red[0] / (float)HD + 1e-6f);
  dst[d] = (bf16_t)(gw[d] * v * r);
}
__global__ __launch_bounds__(256) void softmax_causal_kernel(const float* __restrict__ S,
                                                             bf16_t* __restrict__ P, int T) {
  __shared__ float red[256];
  int r = blockIdx.x;
  const float* sr = S + (size_t)r * T;
  bf16_t* pr = P + (size_t)r * T;
  int n = r + 1;
  float mx = -3.4e38f;
  for (int c = threadIdx.x; c < n; c += 256) mx = fmaxf(mx, sr[c]);
  red[threadIdx.x] = mx; __syncthreads();
  for (int s = 128; s > 0; s >>= 1) {
    if (threadIdx.x < s) red[threadIdx.x] = fmaxf(red[threadIdx.x], red[threadIdx.x + s]);
    __syncthreads();
  }
  mx = red[0]; __syncthreads();
  float sum = 0.f;
  for (int c = threadIdx.x; c < n; c += 256) sum += __expf(sr[c] - mx);
  red[threadIdx.x] = sum; __syncthreads();
  for (int s = 128; s > 0; s >>= 1) {
    if (threadIdx.x < s) red[threadIdx.x] += red[threadIdx.x + s];
    __syncthreads();
  }
  float inv = 1.f / red[0];
  for (int c = threadIdx.x; c < T; c += 256)
    pr[c] = (c < n) ? (bf16_t)(__expf(sr[c] - mx) * inv) : (bf16_t)0.f;
}
__global__ __launch_bounds__(256) void vtranspose_kernel(const float* __restrict__ v,
                                                         bf16_t* __restrict__ vt, int T) {
  size_t i = (size_t)blockIdx.x * 256 + threadIdx.x;   // over T * NKV*HD
  if (i >= (size_t)T * NKV * HD) return;
  int t = (int)(i / (NKV * HD));
  int j = (int)(i % (NKV * HD));
  vt[(size_t)j * T + t] = (bf16_t)v[i];
}
// gate softmax + top-8 + renormalize -> comb[T,16]
__global__ __launch_bounds__(256) void router_kernel(const bf16_t* __restrict__ x2b,
                                                     const float* __restrict__ gw,
                                                     float* __restrict__ comb, int D) {
  __shared__ float red[256];
  __shared__ float logits[NEXP];
  int t = blockIdx.x;
  const bf16_t* xr = x2b + (size_t)t * D;
  for (int e = 0; e < NEXP; ++e) {
    float p = 0.f;
    for (int d = threadIdx.x; d < D; d += 256) p += (float)xr[d] * gw[(size_t)e * D + d];
    red[threadIdx.x] = p; __syncthreads();
    for (int s = 128; s > 0; s >>= 1) {
      if (threadIdx.x < s) red[threadIdx.x] += red[threadIdx.x + s];
      __syncthreads();
    }
    if (threadIdx.x == 0) logits[e] = red[0];
    __syncthreads();
  }
  if (threadIdx.x == 0) {
    float mx = logits[0];
    for (int e = 1; e < NEXP; ++e) mx = fmaxf(mx, logits[e]);
    float g[NEXP]; float gsum = 0.f;
    for (int e = 0; e < NEXP; ++e) { g[e] = __expf(logits[e] - mx); gsum += g[e]; }
    for (int e = 0; e < NEXP; ++e) g[e] /= gsum;
    bool used[NEXP] = {false};
    float w[NEXP] = {0.f}; float tsum = 0.f;
    for (int kk = 0; kk < 8; ++kk) {
      int best = 0; float bv = -1.f;
      for (int e = 0; e < NEXP; ++e)
        if (!used[e] && g[e] > bv) { bv = g[e]; best = e; }
      used[best] = true; w[best] = bv; tsum += bv;
    }
    for (int e = 0; e < NEXP; ++e)
      comb[(size_t)t * NEXP + e] = used[e] ? w[e] / tsum : 0.f;
  }
}

extern "C" void kernel_launch(void* const* d_in, const int* in_sizes, int n_in,
                              void* d_out, int out_size, void* d_ws, size_t ws_size,
                              hipStream_t stream) {
  (void)in_sizes; (void)n_in; (void)out_size; (void)ws_size;
  const float* hidden_in = (const float*)d_in[0];
  // d_in[1] position_ids == arange(T); position == token index, used implicitly.
  const float* cost = (const float*)d_in[2];
  const float* sint = (const float*)d_in[3];
  const float* ln1  = (const float*)d_in[4];
  const float* ln2  = (const float*)d_in[5];
  const float* wq   = (const float*)d_in[6];
  const float* wk   = (const float*)d_in[7];
  const float* wv   = (const float*)d_in[8];
  const float* wo   = (const float*)d_in[9];
  const float* qn   = (const float*)d_in[10];
  const float* kn   = (const float*)d_in[11];
  const float* gw   = (const float*)d_in[12];
  const float* sg   = (const float*)d_in[13];
  const float* su   = (const float*)d_in[14];
  const float* sd   = (const float*)d_in[15];
  const float* eg   = (const float*)d_in[16];
  const float* eu   = (const float*)d_in[17];
  const float* ed   = (const float*)d_in[18];
  float* out = (float*)d_out;

  const int T = T_TOK;
  size_t off = 0;
  auto carve = [&](size_t bytes) -> void* {
    void* p = (char*)d_ws + off;
    off += (bytes + 255) & ~(size_t)255;
    return p;
  };
  bf16_t* xb   = (bf16_t*)carve((size_t)T * DMODEL * 2);   // ln1 out, later reused as x2b
  float*  hid  = (float*)carve((size_t)T * DMODEL * 4);
  float*  qf   = (float*)carve((size_t)T * NH * HD * 4);   // later reused as attn f32
  float*  kf   = (float*)carve((size_t)T * NKV * HD * 4);
  float*  vf   = (float*)carve((size_t)T * NKV * HD * 4);
  bf16_t* qb   = (bf16_t*)carve((size_t)T * NH * HD * 2);
  bf16_t* kb   = (bf16_t*)carve((size_t)T * NKV * HD * 2);
  bf16_t* vt   = (bf16_t*)carve((size_t)NKV * HD * T * 2);
  float*  big1 = (float*)carve((size_t)T * I_SH * 4);      // scores, then gate-proj out
  float*  big2 = (float*)carve((size_t)T * I_SH * 4);      // probs(bf16), then up-proj out
  bf16_t* hbuf = (bf16_t*)carve((size_t)T * I_SH * 2);     // attn bf16, then silu*u bf16
  float*  comb = (float*)carve((size_t)T * NEXP * 4);
  float*  sbuf  = big1;
  bf16_t* probs = (bf16_t*)big2;
  float*  attnf = qf;
  bf16_t* attnb = hbuf;

  dim3 blk(256);
  size_t nTD = (size_t)T * DMODEL;
  unsigned gTD = (unsigned)((nTD + 255) / 256);

  auto gemmF = [&](const bf16_t* A, const float* B, float* C, int M, int N, int K,
                   int lda, int ldb, int ldc, float alpha, const float* rs, int rss, int accum) {
    dim3 g((unsigned)(N / BN), (unsigned)(M / BM));
    gemm_bf16_kernel<float><<<g, blk, 0, stream>>>(A, B, C, M, N, K, lda, ldb, ldc,
                                                   alpha, rs, rss, accum);
  };
  auto gemmB = [&](const bf16_t* A, const bf16_t* B, float* C, int M, int N, int K,
                   int lda, int ldb, int ldc, float alpha, int accum) {
    dim3 g((unsigned)(N / BN), (unsigned)(M / BM));
    gemm_bf16_kernel<bf16_t><<<g, blk, 0, stream>>>(A, B, C, M, N, K, lda, ldb, ldc,
                                                    alpha, nullptr, 0, accum);
  };

  // ---- attention block ----
  copyf_kernel<<<gTD, blk, 0, stream>>>(hidden_in, hid, nTD);           // residual
  rmsnorm_bf16_kernel<<<T, blk, 0, stream>>>(hidden_in, ln1, xb, DMODEL);

  gemmF(xb, wq, qf, T, NH * HD,  DMODEL, DMODEL, DMODEL, NH * HD,  1.f, nullptr, 0, 0);
  gemmF(xb, wk, kf, T, NKV * HD, DMODEL, DMODEL, DMODEL, NKV * HD, 1.f, nullptr, 0, 0);
  gemmF(xb, wv, vf, T, NKV * HD, DMODEL, DMODEL, DMODEL, NKV * HD, 1.f, nullptr, 0, 0);

  rope_qknorm_kernel<<<T * (NH + NKV), dim3(128), 0, stream>>>(qf, kf, cost, sint, qn, kn, qb, kb);
  vtranspose_kernel<<<(unsigned)(((size_t)T * NKV * HD + 255) / 256), blk, 0, stream>>>(vf, vt, T);

  const float scale = 1.f / sqrtf((float)HD);
  for (int h = 0; h < NH; ++h) {
    const bf16_t* qh = qb + h * HD;
    const bf16_t* kh = kb + (h / 4) * HD;
    const bf16_t* vh = vt + (size_t)(h / 4) * HD * T;
    gemmB(qh, kh, sbuf, T, T, HD, NH * HD, NKV * HD, T, scale, 0);
    softmax_causal_kernel<<<T, blk, 0, stream>>>(sbuf, probs, T);
    gemmB(probs, vh, attnf + h * HD, T, HD, T, T, T, NH * HD, 1.f, 0);
  }
  cast_bf16_kernel<<<gTD, blk, 0, stream>>>(attnf, attnb, nTD);
  gemmF(attnb, wo, hid, T, DMODEL, NH * HD, NH * HD, NH * HD, DMODEL, 1.f, nullptr, 0, 1);

  // ---- MoE block ----
  rmsnorm_bf16_kernel<<<T, blk, 0, stream>>>(hid, ln2, xb, DMODEL);     // x2 -> xb (reuse)
  copyf_kernel<<<gTD, blk, 0, stream>>>(hid, out, nTD);                 // out = residual2

  // shared expert
  gemmF(xb, sg, big1, T, I_SH, DMODEL, DMODEL, DMODEL, I_SH, 1.f, nullptr, 0, 0);
  gemmF(xb, su, big2, T, I_SH, DMODEL, DMODEL, DMODEL, I_SH, 1.f, nullptr, 0, 0);
  size_t nSH = (size_t)T * I_SH;
  silu_mul_kernel<<<(unsigned)((nSH + 255) / 256), blk, 0, stream>>>(big1, big2, hbuf, nSH);
  gemmF(hbuf, sd, out, T, DMODEL, I_SH, I_SH, I_SH, DMODEL, 1.f, nullptr, 0, 1);

  // router -> comb[T,16]
  router_kernel<<<T, blk, 0, stream>>>(xb, gw, comb, DMODEL);

  // routed experts (dense over all experts, matching reference)
  size_t nRT = (size_t)T * I_RT;
  for (int e = 0; e < NEXP; ++e) {
    const float* ege = eg + (size_t)e * I_RT * DMODEL;
    const float* eue = eu + (size_t)e * I_RT * DMODEL;
    const float* ede = ed + (size_t)e * DMODEL * I_RT;
    gemmF(xb, ege, big1, T, I_RT, DMODEL, DMODEL, DMODEL, I_RT, 1.f, nullptr, 0, 0);
    gemmF(xb, eue, big2, T, I_RT, DMODEL, DMODEL, DMODEL, I_RT, 1.f, nullptr, 0, 0);
    silu_mul_kernel<<<(unsigned)((nRT + 255) / 256), blk, 0, stream>>>(big1, big2, hbuf, nRT);
    gemmF(hbuf, ede, out, T, DMODEL, I_RT, I_RT, I_RT, DMODEL, 1.f, comb + e, NEXP, 1);
  }
}
